// PointNetSetFeat_88527865905278
// MI455X (gfx1250) — compile-verified, matrix operand order checked by reading, not run on hardware
//
#include <hip/hip_runtime.h>
#include <math.h>

// ---------------------------------------------------------------------------
// DGCNN-style fused point feature network for MI455X (gfx1250, wave32, WMMA)
//   stage A: KNN (k=16) over 8192 points x 4 batches      -> idx in workspace
//   stage B: weight repack fp32 -> bf16 WMMA A-operand    -> workspace
//   stage C: fully fused edge-feature + conv_res x2 + k-max, all GEMMs done
//            with v_wmma_f32_16x16x32_bf16, activations resident in LDS.
// Round-3 tuning:
//   * explicit 2-tile software pipelining with independent accumulators:
//     both WMMA chains issue back-to-back, then both epilogues run, so the
//     WMMA->VALU 4-slot RAW hazard (ISA 7.12.1) is filled with the other
//     tile's matrix work instead of v_nops.
// ---------------------------------------------------------------------------

typedef __attribute__((ext_vector_type(16))) __bf16 v16bf;
typedef __attribute__((ext_vector_type(8)))  __bf16 v8bf;
typedef __attribute__((ext_vector_type(8)))  float  v8f;

#define NPTS   8192
#define NBATCH 4
#define KNN    16

// ---- workspace layout -----------------------------------------------------
#define IDX_BYTES (NBATCH * NPTS * KNN * 4)   // 2 MB of int32 knn indices
// weight-operand offsets (bf16 elements), each matrix: NO*NK*32lanes*16elems
#define OFF_A1   0        // 64x10  -> NO=4 NK=1 : 2048
#define OFF_B1   2048     // 64x64  -> NO=4 NK=2 : 4096
#define OFF_C1   6144     // 64x64  -> NO=4 NK=2 : 4096
#define OFF_SC1  10240    // 64x10  -> NO=4 NK=1 : 2048
#define OFF_A2   12288    // 64x128 -> NO=4 NK=4 : 8192
#define OFF_B2   20480    // 128x64 -> NO=8 NK=2 : 8192
#define OFF_C2   28672    // 128x128-> NO=8 NK=4 : 16384
#define OFF_SC2  45056    // 128x128-> NO=8 NK=4 : 16384

// ---------------------------------------------------------------------------
static __device__ __forceinline__ v8f bfwmma(v16bf a, v16bf b, v8f c) {
  // D(f32 16x16) = A(bf16 16x32) * B(bf16 32x16) + C
  return __builtin_amdgcn_wmma_f32_16x16x32_bf16(false, a, false, b,
                                                 (short)0, c, false, false);
}

// B-operand fragment from LDS activation buffer: chunk kc (32 channels),
// column-tile t (16 columns). lane&15 = column, lane>>4 = K-half.
static __device__ __forceinline__ v16bf bfrag(const __bf16* base, int kc, int t, int lane) {
  return *(const v16bf*)(base + (((kc * 16 + t) * 32) + lane) * 16);
}

// A-operand fragment from pre-packed global weight buffer.
static __device__ __forceinline__ v16bf afrag(const __bf16* w, int nk, int ot, int kc, int lane) {
  return *(const v16bf*)(w + (((ot * nk + kc) * 32) + lane) * 16);
}

// Store a 16x16 f32 D tile (o-tile ot, col-tile t) as bf16 into the next
// layer's B-operand layout. Rows held by lane l are K = (ot&1)*16 + 8*(l>>4)+r,
// which land at elements (ot&1)*8 .. +7 of lane l in chunk ot>>1 -> one
// contiguous 16-byte store per lane.
static __device__ __forceinline__ void stfrag(__bf16* base, int ot, int t, int lane, const v8f& v) {
  v8bf pk;
#pragma unroll
  for (int r = 0; r < 8; ++r) pk[r] = (__bf16)v[r];
  *(v8bf*)(base + ((((ot >> 1) * 16 + t) * 32) + lane) * 16 + (ot & 1) * 8) = pk;
}

// branch-free leaky relu: slope 0.2 > 0  =>  leaky(x) == max(x, 0.2x)
static __device__ __forceinline__ float leaky(float x) {
  return fmaxf(x, 0.2f * x);
}

// ---------------------------------------------------------------------------
// Stage A: brute-force KNN, one thread per query point.
// ---------------------------------------------------------------------------
__global__ __launch_bounds__(256)
void knn_kernel(const float* __restrict__ xyz, int* __restrict__ knn_idx) {
  __shared__ float sx[256], sy[256], sz[256];
  const int b = blockIdx.y;
  const int n = blockIdx.x * 256 + threadIdx.x;
  const float* xb = xyz + (size_t)b * 3 * NPTS;
  const float qx = xb[n], qy = xb[NPTS + n], qz = xb[2 * NPTS + n];

  float bd[KNN];
  int   bi[KNN];
#pragma unroll
  for (int i = 0; i < KNN; ++i) { bd[i] = 3.0e38f; bi[i] = 0; }
  float worst = 3.0e38f;
  int   wslot = 0;

  for (int base = 0; base < NPTS; base += 256) {
    __syncthreads();
    const int j = base + threadIdx.x;
    sx[threadIdx.x] = xb[j];
    sy[threadIdx.x] = xb[NPTS + j];
    sz[threadIdx.x] = xb[2 * NPTS + j];
    __syncthreads();
    for (int u = 0; u < 256; ++u) {
      const float dx = qx - sx[u], dy = qy - sy[u], dz = qz - sz[u];
      const float d2 = dx * dx + dy * dy + dz * dz;
      if (d2 < worst) {
        bd[wslot] = d2; bi[wslot] = base + u;
        worst = bd[0]; wslot = 0;
#pragma unroll
        for (int i = 1; i < KNN; ++i)
          if (bd[i] > worst) { worst = bd[i]; wslot = i; }
      }
    }
  }
  int* op = knn_idx + ((size_t)b * NPTS + n) * KNN;
#pragma unroll
  for (int i = 0; i < KNN; ++i) op[i] = bi[i];
}

// ---------------------------------------------------------------------------
// Stage B: repack fp32 weight (O x Cin, row-major) into bf16 WMMA A-operand
// layout, zero-padding K up to NK*32. grid = NO*NK blocks, 512 threads.
// lane l (0..31), elem e (0..15): K = (e&7) + 8*(l>>4) + 16*(e>>3).
// ---------------------------------------------------------------------------
__global__ __launch_bounds__(512)
void prep_w_kernel(const float* __restrict__ src, __bf16* __restrict__ dst,
                   int O, int Cin, int NK) {
  const int l  = threadIdx.x >> 4;
  const int e  = threadIdx.x & 15;
  const int ot = blockIdx.x / NK;
  const int kc = blockIdx.x % NK;
  const int K   = kc * 32 + (e & 7) + 8 * (l >> 4) + 16 * (e >> 3);
  const int row = ot * 16 + (l & 15);
  const float v = (row < O && K < Cin) ? src[row * Cin + K] : 0.0f;
  dst[((size_t)blockIdx.x * 32 + l) * 16 + e] = (__bf16)v;
}

// ---------------------------------------------------------------------------
// Stage C: fused edge-feature + conv_res(10->64) + kmax/concat +
//          conv_res(128->128) + kmax.  One workgroup per (batch, 16 points).
// Columns: 256 = k(0..15 == col-tile) x nn(0..15).  8 waves.
// ---------------------------------------------------------------------------
__global__ __launch_bounds__(256)
void fused_dgcnn_kernel(const float* __restrict__ xyz,
                        const int*   __restrict__ knn_idx,
                        const __bf16* __restrict__ W,
                        const float* __restrict__ a1b, const float* __restrict__ b1b,
                        const float* __restrict__ c1b, const float* __restrict__ sc1b,
                        const float* __restrict__ a2b, const float* __restrict__ b2b,
                        const float* __restrict__ c2b, const float* __restrict__ sc2b,
                        float* __restrict__ out) {
  extern __shared__ char smem[];
  __bf16* BUF2 = (__bf16*)(smem);            // 64 KB : concat input, 4 chunks
  __bf16* BA   = (__bf16*)(smem + 65536);    // 32 KB : A1 / A2
  __bf16* BF   = (__bf16*)(smem + 98304);    // 16 KB : 32-padded features
  __bf16* BB   = (__bf16*)(smem + 114688);   // 32 KB : B1
  __bf16* B2B  = (__bf16*)(smem + 98304);    // 64 KB : B2 (reuses BF+BB+slack)
  float*  PM   = (float*)(smem + 147456);    // 8 KB  : per-wave partial max
  float*  X1F  = (float*)(smem + 155648);    // 4 KB  : 64x16 channel max
  // total dynamic LDS: 163840 B (2 workgroups fit the 320 KB WGP pool)

  const int tid  = threadIdx.x;
  const int lane = tid & 31;
  const int w    = tid >> 5;
  const int h    = lane >> 4;
  const int b    = blockIdx.y;
  const int n0   = blockIdx.x * 16;

  // ---- phase 0: 10-channel edge feature, zero-padded to 32 channels ----
  {
    const int k = tid >> 4, nn = tid & 15, n = n0 + nn;
    const float* xb = xyz + (size_t)b * 3 * NPTS;
    const float cx = xb[n], cy = xb[NPTS + n], cz = xb[2 * NPTS + n];
    const int j = knn_idx[((size_t)b * NPTS + n) * KNN + k];
    const float nx = xb[j], ny = xb[NPTS + j], nz = xb[2 * NPTS + j];
    const float rx = nx - cx, ry = ny - cy, rz = nz - cz;
    const float d = sqrtf(rx * rx + ry * ry + rz * rz + 1e-12f);
    __bf16* p0 = BF + (k * 32 + nn) * 16;        // lane nn   : ch 0-7 (,16-23)
    p0[0] = (__bf16)cx; p0[1] = (__bf16)cy; p0[2] = (__bf16)cz;
    p0[3] = (__bf16)nx; p0[4] = (__bf16)ny; p0[5] = (__bf16)nz;
    p0[6] = (__bf16)rx; p0[7] = (__bf16)ry;
#pragma unroll
    for (int e = 8; e < 16; ++e) p0[e] = (__bf16)0.0f;
    __bf16* p1 = BF + (k * 32 + nn + 16) * 16;   // lane nn+16: ch 8-15 (,24-31)
    p1[0] = (__bf16)rz; p1[1] = (__bf16)d;
#pragma unroll
    for (int e = 2; e < 16; ++e) p1[e] = (__bf16)0.0f;
  }
  __syncthreads();

  // ---- step 1: A1 = leaky(a1 * F + a1_b)  (64 x 256) ----
  {
    const int ot = w >> 1, tb = (w & 1) * 8;
    const v16bf a0 = afrag(W + OFF_A1, 1, ot, 0, lane);
    float bias[8];
#pragma unroll
    for (int r = 0; r < 8; ++r) bias[r] = a1b[ot * 16 + 8 * h + r];
    for (int ti = 0; ti < 8; ti += 2) {
      const int t0 = tb + ti, t1 = tb + ti + 1;
      v8f acc0 = {}, acc1 = {};
      acc0 = bfwmma(a0, bfrag(BF, 0, t0, lane), acc0);
      acc1 = bfwmma(a0, bfrag(BF, 0, t1, lane), acc1);
      v8f o0, o1;
#pragma unroll
      for (int r = 0; r < 8; ++r) o0[r] = leaky(acc0[r] + bias[r]);
#pragma unroll
      for (int r = 0; r < 8; ++r) o1[r] = leaky(acc1[r] + bias[r]);
      stfrag(BA, ot, t0, lane, o0);
      stfrag(BA, ot, t1, lane, o1);
    }
  }
  __syncthreads();

  // ---- step 2: B1 = leaky(b1 * A1 + b1_b)  (64 x 256) ----
  {
    const int ot = w >> 1, tb = (w & 1) * 8;
    const v16bf w0 = afrag(W + OFF_B1, 2, ot, 0, lane);
    const v16bf w1 = afrag(W + OFF_B1, 2, ot, 1, lane);
    float bias[8];
#pragma unroll
    for (int r = 0; r < 8; ++r) bias[r] = b1b[ot * 16 + 8 * h + r];
    for (int ti = 0; ti < 8; ti += 2) {
      const int t0 = tb + ti, t1 = tb + ti + 1;
      v8f acc0 = {}, acc1 = {};
      acc0 = bfwmma(w0, bfrag(BA, 0, t0, lane), acc0);
      acc1 = bfwmma(w0, bfrag(BA, 0, t1, lane), acc1);
      acc0 = bfwmma(w1, bfrag(BA, 1, t0, lane), acc0);
      acc1 = bfwmma(w1, bfrag(BA, 1, t1, lane), acc1);
      v8f o0, o1;
#pragma unroll
      for (int r = 0; r < 8; ++r) o0[r] = leaky(acc0[r] + bias[r]);
#pragma unroll
      for (int r = 0; r < 8; ++r) o1[r] = leaky(acc1[r] + bias[r]);
      stfrag(BB, ot, t0, lane, o0);
      stfrag(BB, ot, t1, lane, o1);
    }
  }
  __syncthreads();

  // ---- step 3: X = c1*B1 + sc1*F + (c1_b+sc1_b); track max over k ----
  {
    const int ot = w >> 1, tb = (w & 1) * 8;
    const v16bf w0 = afrag(W + OFF_C1, 2, ot, 0, lane);
    const v16bf w1 = afrag(W + OFF_C1, 2, ot, 1, lane);
    const v16bf s0 = afrag(W + OFF_SC1, 1, ot, 0, lane);
    float bias[8];
#pragma unroll
    for (int r = 0; r < 8; ++r) {
      const int ch = ot * 16 + 8 * h + r;
      bias[r] = c1b[ch] + sc1b[ch];
    }
    v8f mx;
#pragma unroll
    for (int r = 0; r < 8; ++r) mx[r] = -3.0e38f;
    for (int ti = 0; ti < 8; ti += 2) {
      const int t0 = tb + ti, t1 = tb + ti + 1;
      v8f acc0 = {}, acc1 = {};
      acc0 = bfwmma(w0, bfrag(BB, 0, t0, lane), acc0);   // conv c path
      acc1 = bfwmma(w0, bfrag(BB, 0, t1, lane), acc1);
      acc0 = bfwmma(w1, bfrag(BB, 1, t0, lane), acc0);
      acc1 = bfwmma(w1, bfrag(BB, 1, t1, lane), acc1);
      acc0 = bfwmma(s0, bfrag(BF, 0, t0, lane), acc0);   // shortcut path
      acc1 = bfwmma(s0, bfrag(BF, 0, t1, lane), acc1);
      v8f o0, o1;
#pragma unroll
      for (int r = 0; r < 8; ++r) { o0[r] = acc0[r] + bias[r]; mx[r] = fmaxf(mx[r], o0[r]); }
#pragma unroll
      for (int r = 0; r < 8; ++r) { o1[r] = acc1[r] + bias[r]; mx[r] = fmaxf(mx[r], o1[r]); }
      stfrag(BUF2, ot, t0, lane, o0);                    // -> chunks 0..1
      stfrag(BUF2, ot, t1, lane, o1);
    }
    float* pw = PM + (w * 32 + lane) * 8;
#pragma unroll
    for (int r = 0; r < 8; ++r) pw[r] = mx[r];
  }
  __syncthreads();

  // combine wave-pair partial maxima -> X1F[64][16]
  for (int u = tid; u < 64 * 16; u += 256) {
    const int ch = u >> 4, nn = u & 15;
    const int ot = ch >> 4, wi = ch & 15, hh = wi >> 3, r = wi & 7;
    const int ln = nn + 16 * hh;
    const float m0 = PM[(((ot * 2) + 0) * 32 + ln) * 8 + r];
    const float m1 = PM[(((ot * 2) + 1) * 32 + ln) * 8 + r];
    X1F[u] = fmaxf(m0, m1);
  }
  __syncthreads();

  // broadcast channel-max into concat channels 64..127 (chunks 2..3 of BUF2)
  {
    int m = tid * 64;
    for (int i = 0; i < 64; ++i, ++m) {
      const int e = m & 15, l = (m >> 4) & 31, t = (m >> 9) & 15, kc2 = m >> 13;
      const int ch = kc2 * 32 + (e & 7) + 8 * (l >> 4) + 16 * (e >> 3);
      BUF2[(((2 + kc2) * 16 + t) * 32 + l) * 16 + e] = (__bf16)X1F[ch * 16 + (l & 15)];
    }
  }
  __syncthreads();

  // ---- step 4: A2 = leaky(a2 * concat + a2_b)  (64 x 256) ----
  {
    const int ot = w >> 1, tb = (w & 1) * 8;
    v16bf wk[4];
#pragma unroll
    for (int kc = 0; kc < 4; ++kc) wk[kc] = afrag(W + OFF_A2, 4, ot, kc, lane);
    float bias[8];
#pragma unroll
    for (int r = 0; r < 8; ++r) bias[r] = a2b[ot * 16 + 8 * h + r];
    for (int ti = 0; ti < 8; ti += 2) {
      const int t0 = tb + ti, t1 = tb + ti + 1;
      v8f acc0 = {}, acc1 = {};
#pragma unroll
      for (int kc = 0; kc < 4; ++kc) {
        acc0 = bfwmma(wk[kc], bfrag(BUF2, kc, t0, lane), acc0);
        acc1 = bfwmma(wk[kc], bfrag(BUF2, kc, t1, lane), acc1);
      }
      v8f o0, o1;
#pragma unroll
      for (int r = 0; r < 8; ++r) o0[r] = leaky(acc0[r] + bias[r]);
#pragma unroll
      for (int r = 0; r < 8; ++r) o1[r] = leaky(acc1[r] + bias[r]);
      stfrag(BA, ot, t0, lane, o0);
      stfrag(BA, ot, t1, lane, o1);
    }
  }
  __syncthreads();

  // ---- step 5: B2 = leaky(b2 * A2 + b2_b)  (128 x 256) ----
  {
    const int ot = w;  // 8 waves, 8 o-tiles
    const v16bf w0 = afrag(W + OFF_B2, 2, ot, 0, lane);
    const v16bf w1 = afrag(W + OFF_B2, 2, ot, 1, lane);
    float bias[8];
#pragma unroll
    for (int r = 0; r < 8; ++r) bias[r] = b2b[ot * 16 + 8 * h + r];
    for (int t = 0; t < 16; t += 2) {
      const int t0 = t, t1 = t + 1;
      v8f acc0 = {}, acc1 = {};
      acc0 = bfwmma(w0, bfrag(BA, 0, t0, lane), acc0);
      acc1 = bfwmma(w0, bfrag(BA, 0, t1, lane), acc1);
      acc0 = bfwmma(w1, bfrag(BA, 1, t0, lane), acc0);
      acc1 = bfwmma(w1, bfrag(BA, 1, t1, lane), acc1);
      v8f o0, o1;
#pragma unroll
      for (int r = 0; r < 8; ++r) o0[r] = leaky(acc0[r] + bias[r]);
#pragma unroll
      for (int r = 0; r < 8; ++r) o1[r] = leaky(acc1[r] + bias[r]);
      stfrag(B2B, ot, t0, lane, o0);
      stfrag(B2B, ot, t1, lane, o1);
    }
  }
  __syncthreads();

  // ---- step 6: OUT = max_k( c2*B2 + sc2*concat ) + (c2_b+sc2_b) ----
  {
    const int ot = w, nn = lane & 15;
    v16bf wc[4], ws[4];
#pragma unroll
    for (int kc = 0; kc < 4; ++kc) {
      wc[kc] = afrag(W + OFF_C2, 4, ot, kc, lane);
      ws[kc] = afrag(W + OFF_SC2, 4, ot, kc, lane);
    }
    v8f mx;
#pragma unroll
    for (int r = 0; r < 8; ++r) mx[r] = -3.0e38f;
    for (int t = 0; t < 16; t += 2) {
      const int t0 = t, t1 = t + 1;
      v8f acc0 = {}, acc1 = {};
#pragma unroll
      for (int kc = 0; kc < 4; ++kc) {
        acc0 = bfwmma(wc[kc], bfrag(B2B, kc, t0, lane), acc0);
        acc1 = bfwmma(wc[kc], bfrag(B2B, kc, t1, lane), acc1);
      }
#pragma unroll
      for (int kc = 0; kc < 4; ++kc) {
        acc0 = bfwmma(ws[kc], bfrag(BUF2, kc, t0, lane), acc0);
        acc1 = bfwmma(ws[kc], bfrag(BUF2, kc, t1, lane), acc1);
      }
#pragma unroll
      for (int r = 0; r < 8; ++r) mx[r] = fmaxf(mx[r], fmaxf(acc0[r], acc1[r]));
    }
#pragma unroll
    for (int r = 0; r < 8; ++r) {
      const int ch = ot * 16 + 8 * h + r;
      out[((size_t)b * 128 + ch) * NPTS + n0 + nn] = mx[r] + c2b[ch] + sc2b[ch];
    }
  }
}

// ---------------------------------------------------------------------------
extern "C" void kernel_launch(void* const* d_in, const int* in_sizes, int n_in,
                              void* d_out, int out_size, void* d_ws, size_t ws_size,
                              hipStream_t stream) {
  const float* xyz  = (const float*)d_in[0];
  const float* sc1w = (const float*)d_in[1];  const float* sc1b = (const float*)d_in[2];
  const float* a1w  = (const float*)d_in[3];  const float* a1b  = (const float*)d_in[4];
  const float* b1w  = (const float*)d_in[5];  const float* b1b  = (const float*)d_in[6];
  const float* c1w  = (const float*)d_in[7];  const float* c1b  = (const float*)d_in[8];
  const float* sc2w = (const float*)d_in[9];  const float* sc2b = (const float*)d_in[10];
  const float* a2w  = (const float*)d_in[11]; const float* a2b  = (const float*)d_in[12];
  const float* b2w  = (const float*)d_in[13]; const float* b2b  = (const float*)d_in[14];
  const float* c2w  = (const float*)d_in[15]; const float* c2b  = (const float*)d_in[16];

  int*    knn_idx = (int*)d_ws;
  __bf16* WB      = (__bf16*)((char*)d_ws + IDX_BYTES);

  // stage B: weight repack (tiny)
  prep_w_kernel<<<dim3(4 * 1), 512, 0, stream>>>(a1w,  WB + OFF_A1,  64, 10, 1);
  prep_w_kernel<<<dim3(4 * 2), 512, 0, stream>>>(b1w,  WB + OFF_B1,  64, 64, 2);
  prep_w_kernel<<<dim3(4 * 2), 512, 0, stream>>>(c1w,  WB + OFF_C1,  64, 64, 2);
  prep_w_kernel<<<dim3(4 * 1), 512, 0, stream>>>(sc1w, WB + OFF_SC1, 64, 10, 1);
  prep_w_kernel<<<dim3(4 * 4), 512, 0, stream>>>(a2w,  WB + OFF_A2,  64, 128, 4);
  prep_w_kernel<<<dim3(8 * 2), 512, 0, stream>>>(b2w,  WB + OFF_B2,  128, 64, 2);
  prep_w_kernel<<<dim3(8 * 4), 512, 0, stream>>>(c2w,  WB + OFF_C2,  128, 128, 4);
  prep_w_kernel<<<dim3(8 * 4), 512, 0, stream>>>(sc2w, WB + OFF_SC2, 128, 128, 4);

  // stage A: KNN
  knn_kernel<<<dim3(NPTS / 256, NBATCH), 256, 0, stream>>>(xyz, knn_idx);

  // stage C: fused WMMA pipeline (160 KB dynamic LDS per workgroup)
  fused_dgcnn_kernel<<<dim3(NPTS / 16, NBATCH), 256, 163840, stream>>>(
      xyz, knn_idx, WB, a1b, b1b, c1b, sc1b, a2b, b2b, c2b, sc2b, (float*)d_out);
}
